// AAFM_73452530696929
// MI455X (gfx1250) — matrix operand order repaired and checked by reading, not compiled, not run on hardware
//
#include <hip/hip_runtime.h>
#include <hip/hip_bf16.h>

// MI455X / gfx1250: wave32, WMMA 16x16x32 f16 with f32 accum.
// Pipeline: [proj+exp (WMMA)] -> [fused expA GEMM num+den + combine (WMMA)].
// Single pass over dist (134 MB), den exchanged via LDS -> ~390 MB total HBM.

typedef __attribute__((ext_vector_type(16))) _Float16 v16h;
typedef __attribute__((ext_vector_type(8)))  _Float16 v8h;
typedef __attribute__((ext_vector_type(8)))  float    v8f;
typedef __attribute__((ext_vector_type(4)))  float    v4f;

#define BATCHN 32
#define NG     1024
#define DM     256
#define NTOT   (BATCHN * NG)
#define LDA    40   // LDS row stride in halves (80B: 16B-aligned rows, conflict-padded)

static __device__ __forceinline__ v16h cat16(v8h lo, v8h hi) {
  return __builtin_shufflevector(lo, hi, 0,1,2,3,4,5,6,7,8,9,10,11,12,13,14,15);
}

// A fragment (16x32 f16, contiguous k order shared with B, not ISA table order): lane holds row (lane&15).
// lanes 0-15: K = 0..15 ; lanes 16-31: K = 16..31 (K = h + 16*(lane>>4))
// -> chunks 2*(lane>>4) and 2*(lane>>4)+1 of the 40-half LDS row.
static __device__ __forceinline__ v16h load_a_frag(const _Float16* As, int wr, int lane) {
  const v8h* p = (const v8h*)(As + (wr * 16 + (lane & 15)) * LDA);
  int hi = lane >> 4;
  return cat16(p[2 * hi], p[2 * hi + 1]);
}

// B fragment (32x16 f16): lane holds column (lane&15); K = h + 16*(lane>>4).
// Bs stored transposed in LDS: Bs[col][kk] -> 2 contiguous 16B reads.
static __device__ __forceinline__ v16h load_b_frag_lds(const _Float16* Bs, int col, int lane) {
  const v8h* p = (const v8h*)(Bs + col * LDA);
  int hi = lane >> 4;
  return cat16(p[2 * hi], p[2 * hi + 1]);
}

// ---------------------------------------------------------------------------
// Kernel 1: fused projections.
//   pass 0 (Wk):  ek  = exp(k)      -> EKT  [b][d][n] f16 (transposed for K2 B-frags)
//   pass 1 (Wv):  ekv = exp(k)*v    -> EKVT [b][d][n] f16
//   pass 2 (Wq):  sq  = sigmoid(q)  -> SQ   [b][n][d] f32
// ---------------------------------------------------------------------------
__global__ void __launch_bounds__(256)
aafm_proj(const float* __restrict__ query, const float* __restrict__ key_,
          const float* __restrict__ value,
          const float* __restrict__ Wq, const float* __restrict__ bq,
          const float* __restrict__ Wk, const float* __restrict__ bk,
          const float* __restrict__ Wv, const float* __restrict__ bv,
          _Float16* __restrict__ EKT, _Float16* __restrict__ EKVT,
          float* __restrict__ SQ)
{
  __shared__ _Float16 As[64 * LDA];     // 64x32 A tile (f16)
  __shared__ _Float16 Bs[256 * LDA];    // W^T chunk, stored [e][kk]
  __shared__ _Float16 EKb[64 * 264];    // exp(k) tile kept for the V pass

  const int tid  = threadIdx.x;
  const int lane = tid & 31;
  const int wave = tid >> 5;
  const int wr   = wave >> 1;           // 0..3 (row group)
  const int wc   = wave & 1;            // 0..1 (col group)
  const int r0   = blockIdx.x * 64;     // global flattened row (b*1024+n)

  for (int p = 0; p < 3; ++p) {
    const float* X    = (p == 0) ? key_ : (p == 1) ? value : query;
    const float* W    = (p == 0) ? Wk   : (p == 1) ? Wv    : Wq;
    const float* bias = (p == 0) ? bk   : (p == 1) ? bv    : bq;

    v8f acc[8] = {};
    for (int k0 = 0; k0 < DM; k0 += 32) {
      __syncthreads();
      { // stage A: 64x32, f32 -> f16
        int r = tid >> 2, c = (tid & 3) * 8;
        const float* s = X + (size_t)(r0 + r) * DM + k0 + c;
        v4f x0 = *(const v4f*)s, x1 = *(const v4f*)(s + 4);
        _Float16* d = As + r * LDA + c;
        #pragma unroll
        for (int i = 0; i < 4; ++i) { d[i] = (_Float16)x0[i]; d[4 + i] = (_Float16)x1[i]; }
      }
      { // stage B: Bs[e][kk] = W[e][k0+kk]  (einsum 'bnd,ed->bne' => B = W^T)
        const float* s = W + (size_t)tid * DM + k0;
        _Float16* d = Bs + tid * LDA;
        #pragma unroll
        for (int i = 0; i < 8; ++i) {
          v4f w = *(const v4f*)(s + i * 4);
          d[i*4+0] = (_Float16)w.x; d[i*4+1] = (_Float16)w.y;
          d[i*4+2] = (_Float16)w.z; d[i*4+3] = (_Float16)w.w;
        }
      }
      __syncthreads();
      v16h a = load_a_frag(As, wr, lane);
      #pragma unroll
      for (int t = 0; t < 8; ++t) {
        v16h bfr = load_b_frag_lds(Bs, wc * 128 + t * 16 + (lane & 15), lane);
        acc[t] = __builtin_amdgcn_wmma_f32_16x16x32_f16(
            false, a, false, bfr, (short)0, acc[t], false, false);
      }
    }
    __syncthreads();

    // epilogue: C layout -> VGPR j: row = j + 8*(lane>>4); col = lane&15 within tile
    #pragma unroll
    for (int t = 0; t < 8; ++t) {
      int col = wc * 128 + t * 16 + (lane & 15);
      float bcol = bias[col];
      #pragma unroll
      for (int j = 0; j < 8; ++j) {
        int rl = wr * 16 + j + 8 * (lane >> 4);   // block-local row 0..63
        int ng = r0 + rl;                         // flattened b*1024+n
        int bb = ng >> 10, nn = ng & (NG - 1);
        float val = acc[t][j] + bcol;
        if (p == 0) {
          float ek = __expf(val);
          EKb[rl * 264 + col] = (_Float16)ek;
          EKT[((size_t)bb * DM + col) * NG + nn] = (_Float16)ek;
        } else if (p == 1) {
          float ek = (float)EKb[rl * 264 + col];
          EKVT[((size_t)bb * DM + col) * NG + nn] = (_Float16)(ek * val);
        } else {
          SQ[(size_t)ng * DM + col] = 1.0f / (1.0f + __expf(-val));
        }
      }
    }
    __syncthreads();
  }
}

// ---------------------------------------------------------------------------
// Kernel 2 (fused): single pass over dist computes BOTH
//   num[b,q,d] = sum_k expA * (exp_K*v)   (waves with wc==0)
//   den[b,q,d] = sum_k expA * exp_K       (waves with wc==1)
// then den is exchanged through LDS and the epilogue writes
//   out = sigmoid(q) * num / (den + 1e-8) directly.
// Each wave: 16 rows x 256 cols = 16 accumulator tiles (128 VGPRs,
// using CDNA5's extended VGPR file via s_set_vgpr_msb).
// ---------------------------------------------------------------------------
__global__ void __launch_bounds__(256)
aafm_attn_fused(const float* __restrict__ dist,
                const _Float16* __restrict__ EKT, const _Float16* __restrict__ EKVT,
                const float* __restrict__ SQ,
                float* __restrict__ out,
                const float* __restrict__ alpha_raw)
{
  __shared__ _Float16 As[64 * LDA];     // expA tile (f16)
  __shared__ float    Den[64 * 264];    // denominator exchange (68 KB)

  const int tid  = threadIdx.x;
  const int lane = tid & 31;
  const int wave = tid >> 5;
  const int wr   = wave >> 1;           // 0..3 row group (16 rows each)
  const int wc   = wave & 1;            // 0: numerator matrix, 1: denominator matrix
  const int r0   = blockIdx.x * 64;
  const int bz   = blockIdx.y;

  const _Float16* Mt = (wc == 0 ? EKVT : EKT) + (size_t)bz * DM * NG;

  float ar    = alpha_raw[0];
  float alpha = log1pf(__expf(ar)) + 1e-6f;   // softplus + eps
  float coef  = alpha * 10.0f;                // log2(1024) = 10

  const float* drow = dist + (size_t)bz * NG * NG;
  const int r = tid >> 2, c = (tid & 3) * 8;
  const float* s0 = drow + (size_t)(r0 + r) * NG + c;

  v8f acc[16] = {};
  for (int k0 = 0; k0 < NG; k0 += 32) {
    __syncthreads();
    { // stage expA tile 64x32 as f16
      const float* s = s0 + k0;
      v4f x0 = *(const v4f*)s, x1 = *(const v4f*)(s + 4);
      __builtin_prefetch((const void*)(s + 32), 0, 0);   // next chunk -> global_prefetch_b8
      _Float16* d = As + r * LDA + c;
      #pragma unroll
      for (int i = 0; i < 4; ++i) {
        d[i]     = (_Float16)__expf(-coef * x0[i]);
        d[4 + i] = (_Float16)__expf(-coef * x1[i]);
      }
    }
    __syncthreads();
    v16h a = load_a_frag(As, wr, lane);
    #pragma unroll
    for (int t = 0; t < 16; ++t) {
      int col = t * 16 + (lane & 15);
      // lane reads K = k0 + 16*(lane>>4) .. +15, contiguous 32B (B-frag layout)
      const v16h* bp = (const v16h*)(Mt + (size_t)col * NG + k0 + ((lane >> 4) << 4));
      v16h bfr = *bp;
      acc[t] = __builtin_amdgcn_wmma_f32_16x16x32_f16(
          false, a, false, bfr, (short)0, acc[t], false, false);
    }
  }

  // --- epilogue: exchange den through LDS, fuse final combine ---
  if (wc == 1) {
    #pragma unroll
    for (int t = 0; t < 16; ++t) {
      int colD = t * 16 + (lane & 15);
      #pragma unroll
      for (int j = 0; j < 8; ++j) {
        int rl = wr * 16 + j + 8 * (lane >> 4);
        Den[rl * 264 + colD] = acc[t][j];
      }
    }
  }
  __syncthreads();
  if (wc == 0) {
    #pragma unroll
    for (int t = 0; t < 16; ++t) {
      int colD = t * 16 + (lane & 15);
      #pragma unroll
      for (int j = 0; j < 8; ++j) {
        int rl = wr * 16 + j + 8 * (lane >> 4);
        int n  = r0 + rl;
        size_t idx = ((size_t)bz * NG + n) * DM + colD;
        float den = Den[rl * 264 + colD];
        float sq  = SQ[idx];
        out[idx]  = sq * acc[t][j] / (den + 1e-8f);
      }
    }
  }
}

extern "C" void kernel_launch(void* const* d_in, const int* in_sizes, int n_in,
                              void* d_out, int out_size, void* d_ws, size_t ws_size,
                              hipStream_t stream) {
  (void)in_sizes; (void)n_in; (void)out_size; (void)ws_size;
  const float* query = (const float*)d_in[0];
  const float* key_  = (const float*)d_in[1];
  const float* value = (const float*)d_in[2];
  const float* dist  = (const float*)d_in[3];
  const float* Wq    = (const float*)d_in[4];
  const float* bq    = (const float*)d_in[5];
  const float* Wk    = (const float*)d_in[6];
  const float* bk    = (const float*)d_in[7];
  const float* Wv    = (const float*)d_in[8];
  const float* bv    = (const float*)d_in[9];
  const float* araw  = (const float*)d_in[10];

  // workspace layout:
  //   EKT  f16 [32][256][1024]  16 MB @ 0
  //   EKVT f16 [32][256][1024]  16 MB @ 16MB
  //   SQ   f32 [32][1024][256]  32 MB @ 32MB   (total 64MB)
  char* ws = (char*)d_ws;
  _Float16* EKT  = (_Float16*)(ws);
  _Float16* EKVT = (_Float16*)(ws + (size_t)16777216);
  float*    SQ   = (float*)   (ws + (size_t)33554432);

  aafm_proj<<<dim3(NTOT / 64), 256, 0, stream>>>(
      query, key_, value, Wq, bq, Wk, bk, Wv, bv, EKT, EKVT, SQ);

  aafm_attn_fused<<<dim3(NG / 64, BATCHN), 256, 0, stream>>>(
      dist, EKT, EKVT, SQ, (float*)d_out, araw);
}
